// RGCNEncoder_49916109914172
// MI455X (gfx1250) — matrix operand order repaired and checked by reading, not compiled, not run on hardware
//
#include <hip/hip_runtime.h>

typedef __attribute__((ext_vector_type(16))) _Float16 v16h;
typedef __attribute__((ext_vector_type(8)))  _Float16 v8h;
typedef __attribute__((ext_vector_type(8)))  float    v8f;

#define H   256
#define R   16
#define NB  8
#define KQ  (NB * H + H)   // 2304 = 8 bases * 256 + 256 root cols

// ---------------- zero workspace (pre + cnt) ----------------
__global__ void zero_f4_kernel(float4* __restrict__ p, long n4) {
    long i = (long)blockIdx.x * blockDim.x + threadIdx.x;
    long stride = (long)gridDim.x * blockDim.x;
    float4 z = {0.f, 0.f, 0.f, 0.f};
    for (; i < n4; i += stride) p[i] = z;
}

// ---------------- per-(dst,rel) edge counts ----------------
__global__ void count_kernel(const int* __restrict__ dst, const int* __restrict__ et,
                             unsigned* __restrict__ cnt, int E) {
    int i = blockIdx.x * blockDim.x + threadIdx.x;
    if (i < E) atomicAdd(&cnt[(size_t)dst[i] * R + et[i]], 1u);
}

// ---------------- scatter x_src rows into pre[dst, rel, :] (one wave per edge) ----------------
__global__ void scatter_kernel(const float* __restrict__ x, const int* __restrict__ srcv,
                               const int* __restrict__ dstv, const int* __restrict__ etv,
                               float* __restrict__ pre, int E) {
    int e    = blockIdx.x * 8 + (threadIdx.x >> 5);
    int lane = threadIdx.x & 31;
    if (e >= E) return;
    int s = srcv[e], d = dstv[e], r = etv[e];
    const float* xp = x + (size_t)s * H;
    float*       pp = pre + ((size_t)d * R + r) * H;
#pragma unroll
    for (int j = 0; j < 8; ++j) {
        int idx = lane + j * 32;
        __hip_atomic_fetch_add(&pp[idx], xp[idx], __ATOMIC_RELAXED, __HIP_MEMORY_SCOPE_AGENT);
    }
}

// ---------------- f32 -> f16 cast (build [basis; root] weight stack) ----------------
__global__ void cast_f16_kernel(const float* __restrict__ a, _Float16* __restrict__ o, long n) {
    long i = (long)blockIdx.x * blockDim.x + threadIdx.x;
    long stride = (long)gridDim.x * blockDim.x;
    for (; i < n; i += stride) o[i] = (_Float16)a[i];
}

// ---------------- fused compress: one pass over pre -> q1 (comp1) AND q2 (comp2) ----------------
// q[dst, b*H+h] = f16( sum_r comp[r,b] * pre[dst,r,h] / max(cnt[dst,r],1) )
// q1 root cols <- f16(x_src[dst,h]); q2 root cols are filled later by gemm1's epilogue.
__global__ void compress_dual_kernel(const float* __restrict__ pre, const unsigned* __restrict__ cnt,
                                     const float* __restrict__ comp1, const float* __restrict__ comp2,
                                     const float* __restrict__ xsrc,
                                     _Float16* __restrict__ q1, _Float16* __restrict__ q2) {
    int d = blockIdx.x;
    int h = threadIdx.x;
    const unsigned* c = cnt + (size_t)d * R;
    float pr[R];
    const float* pp = pre + (size_t)d * R * H + h;
#pragma unroll
    for (int r = 0; r < R; ++r) {
        unsigned cv = c[r];
        float inv = 1.0f / (float)(cv ? cv : 1u);
        pr[r] = pp[(size_t)r * H] * inv;
    }
    _Float16* q1p = q1 + (size_t)d * KQ;
    _Float16* q2p = q2 + (size_t)d * KQ;
#pragma unroll
    for (int b = 0; b < NB; ++b) {
        float s1 = 0.f, s2 = 0.f;
#pragma unroll
        for (int r = 0; r < R; ++r) {
            s1 += comp1[r * NB + b] * pr[r];
            s2 += comp2[r * NB + b] * pr[r];
        }
        q1p[b * H + h] = (_Float16)s1;
        q2p[b * H + h] = (_Float16)s2;
    }
    q1p[NB * H + h] = (_Float16)xsrc[(size_t)d * H + h];
}

// ---------------- fragment loaders ----------------
__device__ __forceinline__ v16h load_a_frag(const _Float16* __restrict__ p) {
    // A (16-bit, 16x32): lane holds K {kh..kh+7} in elems 0..7 and K {kh+16..kh+23} in elems 8..15
    v8h a0 = *(const v8h*)p;
    v8h a1 = *(const v8h*)(p + 16);
    v16h v;
#pragma unroll
    for (int i = 0; i < 8; ++i) { v[i] = a0[i]; v[i + 8] = a1[i]; }
    return v;
}

// ---------------- WMMA GEMM: C[M x 256] = A[M x 2304](f16) * W[2304 x 256](f16), +bias, PReLU.
// Block = 256 threads = 8 waves, 2(M) x 4(N) waves, wave tile 64x16 (4 wmma accumulators).
// Block tile 128 x 64.  Two independent accumulator chains (even/odd 32-K steps) so the
// software pipeline has no loop-carried register copies and no WMMA->VALU hazard NOPs.
// __launch_bounds__(256, 2): cap VGPRs below 256 (no s_set_vgpr_msb traffic) and keep
// >=2 waves/SIMD for inter-wave latency hiding.
// Epilogue writes f32 (stride H) and/or f16 (stride ldoH).
__global__ __launch_bounds__(256, 2) void gemm_kernel(
        const _Float16* __restrict__ A, const _Float16* __restrict__ W,
        const float* __restrict__ bias, const float* __restrict__ pw,
        float* __restrict__ outF, _Float16* __restrict__ outH, int ldoH, int M) {
    int tid   = threadIdx.x;
    int lane  = tid & 31;
    int wid   = tid >> 5;
    int waveM = wid >> 2;          // 0..1
    int waveN = wid & 3;           // 0..3
    int blockM = blockIdx.y * 128;
    int blockN = blockIdx.x * 64;

    int rowA0  = blockM + waveM * 64 + (lane & 15);
    int kHalfA = (lane >> 4) * 8;
    int nB     = blockN + waveN * 16;

    const _Float16* aB[4];
#pragma unroll
    for (int t = 0; t < 4; ++t) {
        int rowA = rowA0 + t * 16;
        if (rowA > M - 1) rowA = M - 1;          // clamp for non-multiple M
        aB[t] = A + (size_t)rowA * KQ + kHalfA;
    }
    const _Float16* bB = W + (size_t)((lane >> 4) * 8) * H + nB + (lane & 15);   // B: lane = N col, elems strided over K (table order)

    v8f acc0[4], acc1[4];
#pragma unroll
    for (int t = 0; t < 4; ++t)
#pragma unroll
        for (int j = 0; j < 8; ++j) { acc0[t][j] = 0.f; acc1[t][j] = 0.f; }

    v16h af0[4], af1[4];
    v16h bf0, bf1;

    auto load_step = [&](int k0, v16h af[4], v16h& bf) {
#pragma unroll
        for (int s = 0; s < 16; ++s) bf[s] = bB[(size_t)(k0 + (s < 8 ? s : s + 8)) * H];
#pragma unroll
        for (int t = 0; t < 4; ++t) af[t] = load_a_frag(aB[t] + k0);
    };
    auto wmma_step = [&](v16h af[4], v16h& bf, v8f acc[4]) {
#pragma unroll
        for (int t = 0; t < 4; ++t)
            acc[t] = __builtin_amdgcn_wmma_f32_16x16x32_f16(
                false, af[t], false, bf, (short)0, acc[t], false, false);
    };

    // Software pipeline: steady state consumes step k0-64 / k0-32 while prefetching k0 / k0+32.
    // KQ = 2304 is a multiple of 64, so exactly (KQ/32) steps are consumed.
    load_step(0,  af0, bf0);
    load_step(32, af1, bf1);
    for (int k0 = 64; k0 < KQ; k0 += 64) {
        wmma_step(af0, bf0, acc0);
        load_step(k0, af0, bf0);
        wmma_step(af1, bf1, acc1);
        load_step(k0 + 32, af1, bf1);
    }
    wmma_step(af0, bf0, acc0);
    wmma_step(af1, bf1, acc1);

    // Epilogue: C/D layout -> lane l, vgpr j: M = 8*(l>>4)+j, N = l&15
    int col = nB + (lane & 15);
    float bi = bias[col];
    float w  = pw[col];
#pragma unroll
    for (int t = 0; t < 4; ++t) {
        int rb = blockM + waveM * 64 + t * 16 + (lane >> 4) * 8;
#pragma unroll
        for (int j = 0; j < 8; ++j) {
            int row = rb + j;
            if (row >= M) continue;
            float v = acc0[t][j] + acc1[t][j] + bi;
            v = (v >= 0.f) ? v : w * v;
            if (outF) outF[(size_t)row * H + col] = v;
            if (outH) outH[(size_t)row * ldoH + col] = (_Float16)v;
        }
    }
}

extern "C" void kernel_launch(void* const* d_in, const int* in_sizes, int n_in,
                              void* d_out, int out_size, void* d_ws, size_t ws_size,
                              hipStream_t stream) {
    const float* x_src   = (const float*)d_in[0];
    const float* basis1  = (const float*)d_in[1];
    const float* comp1   = (const float*)d_in[2];
    const float* root1   = (const float*)d_in[3];
    const float* bias1   = (const float*)d_in[4];
    const float* basis2  = (const float*)d_in[5];
    const float* comp2   = (const float*)d_in[6];
    const float* root2   = (const float*)d_in[7];
    const float* bias2   = (const float*)d_in[8];
    const float* prelu_w = (const float*)d_in[9];
    const int*   e_src   = (const int*)d_in[10];
    const int*   e_dst   = (const int*)d_in[11];
    const int*   e_typ   = (const int*)d_in[12];

    int E   = in_sizes[10];
    int Bsz = out_size / H;          // batch size (8192)

    // ---- workspace carve ----
    char* ws = (char*)d_ws;
    float* pre = (float*)ws;                         // [Bsz, R, H] f32 edge-sum numerators
    ws += (size_t)Bsz * R * H * sizeof(float);
    unsigned* cnt = (unsigned*)ws;                   // [Bsz, R]
    ws += (size_t)Bsz * R * sizeof(unsigned);
    _Float16* q1 = (_Float16*)ws;                    // [Bsz, KQ]
    ws += (size_t)Bsz * KQ * sizeof(_Float16);
    _Float16* q2 = (_Float16*)ws;                    // [Bsz, KQ]
    ws += (size_t)Bsz * KQ * sizeof(_Float16);
    _Float16* w1 = (_Float16*)ws;                    // [KQ, H] = [basis1; root1] f16
    ws += (size_t)KQ * H * sizeof(_Float16);
    _Float16* w2 = (_Float16*)ws;                    // [KQ, H] = [basis2; root2] f16

    // 1) zero pre + cnt (adjacent, both 16B multiples)
    long zero4 = ((long)Bsz * R * H * 4 + (long)Bsz * R * 4) / 16;
    zero_f4_kernel<<<4096, 256, 0, stream>>>((float4*)pre, zero4);

    // 2) counts, 3) scatter (shared by both layers: all dst < Bsz, sources are original x_src)
    count_kernel<<<(E + 255) / 256, 256, 0, stream>>>(e_dst, e_typ, cnt, E);
    scatter_kernel<<<(E + 7) / 8, 256, 0, stream>>>(x_src, e_src, e_dst, e_typ, pre, E);

    // 4) build f16 weight stacks [basis; root]
    long nb_elems = (long)NB * H * H;   // 524288
    long rt_elems = (long)H * H;        // 65536
    cast_f16_kernel<<<512, 256, 0, stream>>>(basis1, w1, nb_elems);
    cast_f16_kernel<<<128, 256, 0, stream>>>(root1,  w1 + nb_elems, rt_elems);
    cast_f16_kernel<<<512, 256, 0, stream>>>(basis2, w2, nb_elems);
    cast_f16_kernel<<<128, 256, 0, stream>>>(root2,  w2 + nb_elems, rt_elems);

    // 5) single pass over pre -> q1 (comp1 + root cols from x_src) and q2 (comp2)
    compress_dual_kernel<<<Bsz, H, 0, stream>>>(pre, cnt, comp1, comp2, x_src, q1, q2);

    // 6) layer 1 GEMM: h1 = prelu(q1 @ w1 + bias1); write f16 h1 into q2 root cols
    dim3 ggrid(H / 64, (Bsz + 127) / 128);
    gemm_kernel<<<ggrid, 256, 0, stream>>>(q1, w1, bias1, prelu_w,
                                           nullptr, q2 + NB * H, KQ, Bsz);

    // 7) layer 2 GEMM: out = prelu(q2 @ w2 + bias2) -> d_out (f32)
    gemm_kernel<<<ggrid, 256, 0, stream>>>(q2, w2, bias2, prelu_w,
                                           (float*)d_out, nullptr, 0, Bsz);
}
